// FasterRCNN_38817914422174
// MI455X (gfx1250) — matrix operand order repaired and checked by reading, not compiled, not run on hardware
//
#include <hip/hip_runtime.h>
#include <hip/hip_bf16.h>
#include <math.h>

// ---------------------------------------------------------------------------
// Faster-RCNN head for MI455X (gfx1250, wave32).
// Heavy math (512x4608 @ 4608x1024 and 512x1024 @ 1024x1024) runs on
// v_wmma_f32_16x16x32_f16. Everything else is bandwidth-trivial glue.
// ---------------------------------------------------------------------------

typedef _Float16 v8h  __attribute__((ext_vector_type(8)));
typedef _Float16 v16h __attribute__((ext_vector_type(16)));
typedef float    v8f  __attribute__((ext_vector_type(8)));

#define HF 128
#define WF 128
#define DF 512
#define KA 4        // anchors per cell
#define NANCH (HF*WF*KA)
#define MAX_DET 512
#define HIDDEN 1024
#define POOLED_K (3*3*DF)   // 4608
#define TAU 0.5f
#define NMS_IOU 0.3f

// ------------------------- phase 1: RPN decode -----------------------------
__global__ void score_box_kernel(const float* __restrict__ rpn_obj,
                                 const float* __restrict__ rpn_reg,
                                 const float* __restrict__ anchors,
                                 float* __restrict__ work,     // masked scores
                                 float* __restrict__ boxes)    // [N][4] xywh
{
    int n = blockIdx.x * blockDim.x + threadIdx.x;
    if (n >= NANCH) return;
    int k    = n & (KA - 1);
    int cell = n >> 2;                       // i*WF + j
    float o0 = rpn_obj[cell * (2*KA) + k*2 + 0];
    float o1 = rpn_obj[cell * (2*KA) + k*2 + 1];
    float score = 1.0f / (1.0f + expf(o0 - o1));   // softmax[:,1]
    work[n] = (score > TAU) ? score : -INFINITY;

    float ax = anchors[n*4+0], ay = anchors[n*4+1];
    float aw = anchors[n*4+2], ah = anchors[n*4+3];
    float r0 = rpn_reg[cell * (4*KA) + k*4 + 0];
    float r1 = rpn_reg[cell * (4*KA) + k*4 + 1];
    float r2 = rpn_reg[cell * (4*KA) + k*4 + 2];
    float r3 = rpn_reg[cell * (4*KA) + k*4 + 3];
    float cx = ax + aw*0.5f + r0*aw;
    float cy = ay + ah*0.5f + r1*ah;
    float w  = aw * expf(r2);
    float h  = ah * expf(r3);
    boxes[n*4+0] = cx - w*0.5f;
    boxes[n*4+1] = cy - h*0.5f;
    boxes[n*4+2] = w;
    boxes[n*4+3] = h;
}

// ------------------------- phase 2: top-k 512 ------------------------------
__device__ __forceinline__ unsigned int fkey(float f) {
    unsigned int u = __float_as_uint(f);
    return (u & 0x80000000u) ? ~u : (u | 0x80000000u);
}

__global__ void topk_kernel(float* __restrict__ work,
                            const float* __restrict__ boxes,
                            float* __restrict__ topScore,
                            int*   __restrict__ topIdx,
                            float* __restrict__ sel)
{
    __shared__ unsigned long long red[1024];
    const int tid = threadIdx.x;
    for (int it = 0; it < MAX_DET; ++it) {
        unsigned long long best = 0ull;
        for (int n = tid; n < NANCH; n += 1024) {
            float s = work[n];
            if (s == s) {   // NaN marks consumed entries
                unsigned long long key =
                    ((unsigned long long)fkey(s) << 32) | (unsigned int)(~n);
                if (key > best) best = key;
            }
        }
        red[tid] = best;
        __syncthreads();
        for (int st = 512; st > 0; st >>= 1) {
            if (tid < st) {
                if (red[tid + st] > red[tid]) red[tid] = red[tid + st];
            }
            __syncthreads();
        }
        if (tid == 0) {
            unsigned long long kk = red[0];
            int idx = (int)(~(unsigned int)kk);
            topIdx[it]   = idx;
            topScore[it] = work[idx];
            sel[it*4+0] = boxes[idx*4+0];
            sel[it*4+1] = boxes[idx*4+1];
            sel[it*4+2] = boxes[idx*4+2];
            sel[it*4+3] = boxes[idx*4+3];
            work[idx] = __uint_as_float(0x7FC00000u);   // consume
        }
        __syncthreads();
    }
}

// ------------------------- phase 3: NMS ------------------------------------
__global__ void nms_kernel(const float* __restrict__ sel,
                           const float* __restrict__ topScore,
                           float* __restrict__ keep)
{
    __shared__ float X[MAX_DET], Y[MAX_DET], W[MAX_DET], H[MAX_DET];
    __shared__ unsigned char kp[MAX_DET];
    __shared__ int supp;
    const int t = threadIdx.x;
    X[t] = sel[t*4+0]; Y[t] = sel[t*4+1];
    W[t] = sel[t*4+2]; H[t] = sel[t*4+3];
    float s = topScore[t];
    bool valid = (s == s) && (s > -INFINITY) && (s < INFINITY);
    kp[t] = 0;
    __syncthreads();
    for (int i = 0; i < MAX_DET; ++i) {
        if (t == 0) supp = 0;
        __syncthreads();
        if (t < i && kp[t]) {
            float x1a = X[t], y1a = Y[t], x2a = x1a + W[t], y2a = y1a + H[t];
            float x1b = X[i], y1b = Y[i], x2b = x1b + W[i], y2b = y1b + H[i];
            float ix = fmaxf(0.0f, fminf(x2a, x2b) - fmaxf(x1a, x1b));
            float iy = fmaxf(0.0f, fminf(y2a, y2b) - fmaxf(y1a, y1b));
            float inter = ix * iy;
            float iou = inter / (W[t]*H[t] + W[i]*H[i] - inter + 1e-8f);
            if (iou > NMS_IOU) atomicOr(&supp, 1);
        }
        __syncthreads();
        if (t == i) kp[i] = (valid && supp == 0) ? 1 : 0;
        __syncthreads();
    }
    keep[t] = kp[t] ? 1.0f : 0.0f;
}

// ------------------------- phase 4: ROI align ------------------------------
// pooled output stored f16 row-major [512][4608] -> feeds WMMA A-fragments.
__global__ void roialign_kernel(const float* __restrict__ feat,
                                const float* __restrict__ sel,
                                _Float16* __restrict__ pooled)
{
    const int b  = blockIdx.x;
    const int ty = blockIdx.y;
    float bx = sel[b*4+0], by = sel[b*4+1], bw = sel[b*4+2], bh = sel[b*4+3];
    int x0[6], x1[6], y0[6], y1[6];
    float wx[6], wy[6];
#pragma unroll
    for (int s = 0; s < 6; ++s) {
        float tc = (s + 0.5f) / 6.0f;
        float fx = (bx + tc * bw) / 16.0f - 0.5f;
        float fy = (by + tc * bh) / 16.0f - 0.5f;
        fx = fminf(fmaxf(fx, 0.0f), (float)(WF - 1));
        fy = fminf(fmaxf(fy, 0.0f), (float)(HF - 1));
        x0[s] = (int)floorf(fx); x1[s] = min(x0[s] + 1, WF - 1); wx[s] = fx - (float)x0[s];
        y0[s] = (int)floorf(fy); y1[s] = min(y0[s] + 1, HF - 1); wy[s] = fy - (float)y0[s];
    }
    for (int tx = 0; tx < 3; ++tx) {
        for (int c = threadIdx.x; c < DF; c += blockDim.x) {
            float m = -INFINITY;
#pragma unroll
            for (int py = 0; py < 2; ++py) {
#pragma unroll
                for (int px = 0; px < 2; ++px) {
                    int sy = ty*2 + py, sx = tx*2 + px;
                    const float* f00 = feat + ((size_t)(y0[sy]*WF + x0[sx])) * DF;
                    const float* f01 = feat + ((size_t)(y0[sy]*WF + x1[sx])) * DF;
                    const float* f10 = feat + ((size_t)(y1[sy]*WF + x0[sx])) * DF;
                    const float* f11 = feat + ((size_t)(y1[sy]*WF + x1[sx])) * DF;
                    float w0y = 1.0f - wy[sy], w1y = wy[sy];
                    float w0x = 1.0f - wx[sx], w1x = wx[sx];
                    float v = f00[c]*w0y*w0x + f01[c]*w0y*w1x
                            + f10[c]*w1y*w0x + f11[c]*w1y*w1x;
                    m = fmaxf(m, v);
                }
            }
            pooled[(size_t)b * POOLED_K + (ty*3 + tx) * DF + c] = (_Float16)m;
        }
    }
}

// --------------- weight pre-pack into WMMA B-fragment layout ---------------
// Packed index = ((nt*KT + kt)*32 + lane)*16 + t
//   lanes 0-15 : K = kt*32 + t        (t = 0..15), N = nt*16 + (lane&15)
//   lanes 16-31: K = kt*32 + 16 + t
__global__ void pack_w_kernel(const float* __restrict__ W,
                              _Float16* __restrict__ Wp,
                              int K, int N)
{
    int p = blockIdx.x * blockDim.x + threadIdx.x;
    if (p >= K * N) return;
    const int KT = K >> 5;
    int t    = p & 15;
    int lane = (p >> 4) & 31;
    int r    = p >> 9;
    int kt   = r % KT;
    int nt   = r / KT;
    int krow = kt*32 + ((lane >> 4) << 4) + t;
    int col  = nt*16 + (lane & 15);
    Wp[p] = (_Float16)W[(size_t)krow * N + col];
}

// --------------- WMMA GEMM: C = relu(A @ B + bias), f16 out ----------------
// One wave per 16x16 output tile; K-loop of v_wmma_f32_16x16x32_f16.
__global__ void gemm_relu_wmma(const _Float16* __restrict__ A,   // MxK row-major
                               const _Float16* __restrict__ Bp,  // packed
                               const float*    __restrict__ bias,
                               _Float16*       __restrict__ C,   // MxN row-major
                               int M, int N, int K)
{
    const int lane = threadIdx.x & 31;
    const int wave = threadIdx.x >> 5;
    const int tile = blockIdx.x * (blockDim.x >> 5) + wave;
    const int ntn  = N >> 4;
    if (tile >= (M >> 4) * ntn) return;      // wave-uniform: EXEC stays all-1
    const int mt = tile / ntn;
    const int nt = tile - mt * ntn;
    const int KT = K >> 5;

    const int row = (mt << 4) + (lane & 15);
    // A-fragment: lanes 0-15 hold K {0..15}, 16-31 {16..31} (matches packed B)
    const _Float16* a_lo = A + (size_t)row * K + ((lane >> 4) << 4);
    const _Float16* bptr = Bp + ((size_t)nt * KT * 32 + lane) * 16;

    v8f acc = {};
    for (int kt = 0; kt < KT; ++kt) {
        union { v16h v; v8h h[2]; } a, b;
        a.h[0] = *reinterpret_cast<const v8h*>(a_lo + (size_t)kt * 32);
        a.h[1] = *reinterpret_cast<const v8h*>(a_lo + (size_t)kt * 32 + 8);
        b.h[0] = *reinterpret_cast<const v8h*>(bptr + (size_t)kt * 512);
        b.h[1] = *reinterpret_cast<const v8h*>(bptr + (size_t)kt * 512 + 8);
        acc = __builtin_amdgcn_wmma_f32_16x16x32_f16(
                  false, a.v, false, b.v, (short)0, acc, false, false);
    }

    const int n     = (nt << 4) + (lane & 15);
    const float bv  = bias[n];
    const int mbase = (mt << 4) + ((lane >> 4) << 3);
#pragma unroll
    for (int v = 0; v < 8; ++v) {
        float x = fmaxf(acc[v] + bv, 0.0f);
        C[(size_t)(mbase + v) * N + n] = (_Float16)x;
    }
}

// ------------------------- FC3 (N=4, tiny) ---------------------------------
__global__ void fc3_kernel(const _Float16* __restrict__ H,
                           const float* __restrict__ W3,
                           const float* __restrict__ b3,
                           float* __restrict__ deltas)
{
    int idx = blockIdx.x * blockDim.x + threadIdx.x;
    if (idx >= MAX_DET * 4) return;
    int m = idx >> 2, j = idx & 3;
    float acc = b3[j];
    for (int k = 0; k < HIDDEN; ++k)
        acc += (float)H[(size_t)m * HIDDEN + k] * W3[k*4 + j];
    deltas[idx] = acc;
}

// ------------------------- finalize ----------------------------------------
__global__ void finalize_kernel(const float* __restrict__ deltas,
                                const float* __restrict__ sel,
                                const float* __restrict__ topScore,
                                const float* __restrict__ keep,
                                float* __restrict__ out)
{
    int i = blockIdx.x * blockDim.x + threadIdx.x;
    if (i >= MAX_DET) return;
    float ax = sel[i*4+0], ay = sel[i*4+1], aw = sel[i*4+2], ah = sel[i*4+3];
    float d0 = deltas[i*4+0], d1 = deltas[i*4+1];
    float d2 = deltas[i*4+2], d3 = deltas[i*4+3];
    float cx = ax + aw*0.5f + d0*aw;
    float cy = ay + ah*0.5f + d1*ah;
    float w  = aw * expf(d2);
    float h  = ah * expf(d3);
    float s  = topScore[i];
    bool valid = (s == s) && (s > -INFINITY) && (s < INFINITY);
    float sc = valid ? s : 0.0f;
    float k  = keep[i];
    out[i*5+0] = (cx - w*0.5f) * k;
    out[i*5+1] = (cy - h*0.5f) * k;
    out[i*5+2] = w * k;
    out[i*5+3] = h * k;
    out[i*5+4] = sc * k;
}

// ---------------------------------------------------------------------------
extern "C" void kernel_launch(void* const* d_in, const int* in_sizes, int n_in,
                              void* d_out, int out_size, void* d_ws, size_t ws_size,
                              hipStream_t stream)
{
    (void)in_sizes; (void)n_in; (void)out_size;
    const float* features = (const float*)d_in[0];
    const float* rpn_obj  = (const float*)d_in[1];
    const float* rpn_reg  = (const float*)d_in[2];
    const float* anchors  = (const float*)d_in[3];
    const float* W1       = (const float*)d_in[4];
    const float* b1       = (const float*)d_in[5];
    const float* W2       = (const float*)d_in[6];
    const float* b2       = (const float*)d_in[7];
    const float* W3       = (const float*)d_in[8];
    const float* b3       = (const float*)d_in[9];

    char* ws = (char*)d_ws;
    size_t off = 0;
    auto take = [&](size_t bytes) -> char* {
        char* p = ws + off;
        off = (off + bytes + 255) & ~(size_t)255;
        return p;
    };
    float*    work     = (float*)take((size_t)NANCH * 4);
    float*    boxes    = (float*)take((size_t)NANCH * 16);
    float*    topScore = (float*)take(MAX_DET * 4);
    int*      topIdx   = (int*)  take(MAX_DET * 4);
    float*    sel      = (float*)take(MAX_DET * 16);
    float*    keepArr  = (float*)take(MAX_DET * 4);
    _Float16* pooled   = (_Float16*)take((size_t)MAX_DET * POOLED_K * 2);
    _Float16* W1p      = (_Float16*)take((size_t)POOLED_K * HIDDEN * 2);
    _Float16* W2p      = (_Float16*)take((size_t)HIDDEN * HIDDEN * 2);
    _Float16* H1       = (_Float16*)take((size_t)MAX_DET * HIDDEN * 2);
    _Float16* H2       = (_Float16*)take((size_t)MAX_DET * HIDDEN * 2);
    float*    deltas   = (float*)take(MAX_DET * 16);
    if (off > ws_size) return;   // workspace too small; bail deterministically

    score_box_kernel<<<NANCH/256, 256, 0, stream>>>(rpn_obj, rpn_reg, anchors,
                                                    work, boxes);
    pack_w_kernel<<<(POOLED_K*HIDDEN)/256, 256, 0, stream>>>(W1, W1p, POOLED_K, HIDDEN);
    pack_w_kernel<<<(HIDDEN*HIDDEN)/256, 256, 0, stream>>>(W2, W2p, HIDDEN, HIDDEN);
    topk_kernel<<<1, 1024, 0, stream>>>(work, boxes, topScore, topIdx, sel);
    nms_kernel<<<1, MAX_DET, 0, stream>>>(sel, topScore, keepArr);
    roialign_kernel<<<dim3(MAX_DET, 3), 256, 0, stream>>>(features, sel, pooled);

    // 512x1024 output tiles -> 2048 waves -> 256 blocks of 8 waves each
    gemm_relu_wmma<<<256, 256, 0, stream>>>(pooled, W1p, b1, H1,
                                            MAX_DET, HIDDEN, POOLED_K);
    gemm_relu_wmma<<<256, 256, 0, stream>>>(H1, W2p, b2, H2,
                                            MAX_DET, HIDDEN, HIDDEN);
    fc3_kernel<<<(MAX_DET*4)/256, 256, 0, stream>>>(H2, W3, b3, deltas);
    finalize_kernel<<<MAX_DET/256, 256, 0, stream>>>(deltas, sel, topScore,
                                                     keepArr, (float*)d_out);
}
